// PRCutBatchLoss_18425409699883
// MI455X (gfx1250) — matrix operand order repaired and checked by reading, not compiled, not run on hardware
//
#include <hip/hip_runtime.h>
#include <hip/hip_bf16.h>

// ---------------------------------------------------------------------------
// PRCutBatchLoss on MI455X (gfx1250, wave32, WMMA bf16 16x16x32)
//
//   M      = W @ p_r                        (8192x256, f32 accum, bf16 inputs)
//   row_w  = W.sum(axis=1)                  (f32, accumulated while loading W)
//   cs[k]  = sum_a M[a,k]                   (replaces col_w @ p_r)
//   cross  = sum_a p_l[a,k] * M[a,k]
//   rs[k]  = sum_a row_w[a] * p_l[a,k]
//   out    = sum_k (rs + cs - 2*cross) / clusters_p[k]
//
// B tiles are staged into a double-buffered LDS region with
// GLOBAL_LOAD_ASYNC_TO_LDS_B128 (ASYNCcnt path), overlapping the copy of
// K-step k+1 with the 16 WMMAs of K-step k. One barrier per iteration.
// ---------------------------------------------------------------------------

typedef __attribute__((ext_vector_type(16))) __bf16 v16bf;
typedef __attribute__((ext_vector_type(8)))  float  v8f;

#define A_DIM 8192
#define B_DIM 8192
#define K_CL  256
#define BM    128      // rows per workgroup (8 waves x 16)
#define KSPLIT 4       // K-dimension split for occupancy (256 workgroups)
#define KCHUNK (B_DIM / KSPLIT)   // 2048
#define KSTEP 32       // K per WMMA

__device__ __forceinline__ unsigned short f32_to_bf16_rne(float f) {
  unsigned u = __float_as_uint(f);
  u += 0x7FFFu + ((u >> 16) & 1u);      // round-to-nearest-even
  return (unsigned short)(u >> 16);
}

// ---------------------------------------------------------------------------
// Kernel 1: p_r (8192x256 f32, row-major) -> p_rT (256x8192 bf16, row-major)
// so GEMM B-fragments are contiguous 32B per lane.
// ---------------------------------------------------------------------------
__global__ __launch_bounds__(256) void k_transpose_pr(
    const float* __restrict__ pr, unsigned short* __restrict__ prT) {
  __shared__ float tile[32][33];
  const int b0 = blockIdx.x * 32;   // b (row of p_r) tile base
  const int n0 = blockIdx.y * 32;   // cluster tile base
  const int tx = threadIdx.x;       // 32
  const int ty = threadIdx.y;       // 8
#pragma unroll
  for (int j = 0; j < 32; j += 8)
    tile[ty + j][tx] = pr[(size_t)(b0 + ty + j) * K_CL + n0 + tx];
  __syncthreads();
#pragma unroll
  for (int j = 0; j < 32; j += 8)
    prT[(size_t)(n0 + ty + j) * B_DIM + b0 + tx] = f32_to_bf16_rne(tile[tx][ty + j]);
}

// ---------------------------------------------------------------------------
// Kernel 2: M += W[:, kchunk] @ p_r[kchunk, :]  and row_w += W row sums.
// grid (A_DIM/BM, KSPLIT), block 256 (8 waves). Each wave owns 16 rows (A)
// and all 256 clusters (16 accumulator tiles of v8f).
// ---------------------------------------------------------------------------
__global__ __launch_bounds__(256, 1) void k_gemm_wpr(
    const float* __restrict__ W, const unsigned short* __restrict__ prT,
    float* __restrict__ Mmat, float* __restrict__ rowW) {
  const int lane = threadIdx.x & 31;
  const int wave = threadIdx.x >> 5;
  const int rowbase = blockIdx.x * BM + wave * 16;
  const int kstart  = blockIdx.y * KCHUNK;

  // B staging, double buffered: 2 x (32 K x 256 N) bf16 = 2 x 16 KB.
  __shared__ unsigned short ldsB[2][K_CL * 32];

  // async stage of one 16 KB B tile: thread t copies prT[t][k0..k0+31] (64 B)
  // straight into LDS via GLOBAL_LOAD_ASYNC_TO_LDS_B128 (4 x 16 B).
  auto stageB = [&](int k0, int buf) {
    unsigned ldsAddr = (unsigned)(uintptr_t)&ldsB[buf][threadIdx.x * 32];
    unsigned long long ga =
        (unsigned long long)(uintptr_t)(prT + (size_t)threadIdx.x * B_DIM + kstart + k0);
    asm volatile(
        "global_load_async_to_lds_b128 %0, %1, off\n\t"
        "global_load_async_to_lds_b128 %0, %1, off offset:16\n\t"
        "global_load_async_to_lds_b128 %0, %1, off offset:32\n\t"
        "global_load_async_to_lds_b128 %0, %1, off offset:48"
        :: "v"(ldsAddr), "v"(ga) : "memory");
  };

  v8f acc[16];
#pragma unroll
  for (int i = 0; i < 16; ++i) acc[i] = (v8f){0.f,0.f,0.f,0.f,0.f,0.f,0.f,0.f};

  const int half = (lane < 16) ? 0 : 1;          // which K-half this lane holds
  const int arow = rowbase + (lane & 15);        // A row for this lane
  const int aoff = half ? 16 : 0;                // K run start (contiguous A layout)
  const float* wrow = W + (size_t)arow * B_DIM + kstart;
  float rowsum = 0.f;

  stageB(0, 0);                                  // prologue: fill buffer 0

  for (int k0 = 0; k0 < KCHUNK; k0 += KSTEP) {
    const int buf = (k0 >> 5) & 1;

    // ---- A fragment: lane holds K run [aoff..aoff+15]
    const float4* ap = (const float4*)(wrow + k0 + aoff);
    float4 x0 = ap[0], x1 = ap[1];
    const float4* ap2 = (const float4*)(wrow + k0 + aoff + 8);
    float4 y0 = ap2[0], y1 = ap2[1];
    __builtin_prefetch(wrow + k0 + KSTEP, 0, 1);   // global_prefetch_b8 next tile

    rowsum += x0.x + x0.y + x0.z + x0.w + x1.x + x1.y + x1.z + x1.w
            + y0.x + y0.y + y0.z + y0.w + y1.x + y1.y + y1.z + y1.w;

    v16bf av;
    av[0]  = (__bf16)x0.x; av[1]  = (__bf16)x0.y;
    av[2]  = (__bf16)x0.z; av[3]  = (__bf16)x0.w;
    av[4]  = (__bf16)x1.x; av[5]  = (__bf16)x1.y;
    av[6]  = (__bf16)x1.z; av[7]  = (__bf16)x1.w;
    av[8]  = (__bf16)y0.x; av[9]  = (__bf16)y0.y;
    av[10] = (__bf16)y0.z; av[11] = (__bf16)y0.w;
    av[12] = (__bf16)y1.x; av[13] = (__bf16)y1.y;
    av[14] = (__bf16)y1.z; av[15] = (__bf16)y1.w;

    // ---- wait for this iteration's B tile (issued last iteration) to land,
    //      then rendezvous; after the barrier buffer `buf` is valid everywhere
    //      and buffer `buf^1` is no longer being read by anyone.
    asm volatile("s_wait_asynccnt 0" ::: "memory");
    __syncthreads();

    if (k0 + KSTEP < KCHUNK) stageB(k0 + KSTEP, buf ^ 1);  // overlap next copy

    // ---- 16 WMMAs: same A, 16 N-tiles of B. B-lane layout: N = lane&15,
    //      lanes 0-15 hold K 0..15, lanes 16-31 hold K 16..31 (contiguous).
#pragma unroll
    for (int nt = 0; nt < 16; ++nt) {
      const v16bf bv =
          *(const v16bf*)(&ldsB[buf][(nt * 16 + (lane & 15)) * 32 + half * 16]);
      acc[nt] = __builtin_amdgcn_wmma_f32_16x16x32_bf16(
          false, av, false, bv, (short)0, acc[nt], false, false);
    }
  }

  // ---- row sums: lane L and L+16 hold the two K-half partials of row L
  float other = __shfl_down(rowsum, 16, 32);
  if (lane < 16) atomicAdd(&rowW[arow], rowsum + other);

  // ---- C/D layout: VGPR r -> M = r + (lane<16 ? 0 : 8), N = lane&15
  const int mb = half ? 8 : 0;
#pragma unroll
  for (int nt = 0; nt < 16; ++nt) {
    const int kcol = nt * 16 + (lane & 15);
#pragma unroll
    for (int r = 0; r < 8; ++r) {
      const int a = rowbase + mb + r;
      atomicAdd(&Mmat[(size_t)a * K_CL + kcol], acc[nt][r]);
    }
  }
}

// ---------------------------------------------------------------------------
// Kernel 3: out += sum_k ( rs[k] + cs[k] - 2*cross[k] ) / clusters_p[k]
// thread k in [0,256) walks 128 rows, fully coalesced.
// ---------------------------------------------------------------------------
__global__ __launch_bounds__(256) void k_finalize(
    const float* __restrict__ pL, const float* __restrict__ Mmat,
    const float* __restrict__ rowW, const float* __restrict__ cp,
    float* __restrict__ out) {
  const int k  = threadIdx.x;
  const int a0 = blockIdx.x * 128;
  float cr = 0.f, cs = 0.f, rs = 0.f;
  for (int i = 0; i < 128; ++i) {
    const int a = a0 + i;
    const float m  = Mmat[(size_t)a * K_CL + k];
    const float pl = pL[(size_t)a * K_CL + k];
    cr += pl * m;
    cs += m;
    rs += rowW[a] * pl;
  }
  float val = (rs + cs - 2.f * cr) / cp[k];
  __shared__ float red[256];
  red[k] = val;
  __syncthreads();
  for (int s = 128; s > 0; s >>= 1) {
    if (k < s) red[k] += red[k + s];
    __syncthreads();
  }
  if (k == 0) atomicAdd(out, red[0]);
}

// ---------------------------------------------------------------------------
extern "C" void kernel_launch(void* const* d_in, const int* in_sizes, int n_in,
                              void* d_out, int out_size, void* d_ws, size_t ws_size,
                              hipStream_t stream) {
  const float* W  = (const float*)d_in[0];   // (8192, 8192)
  const float* pL = (const float*)d_in[1];   // (8192, 256)
  const float* pR = (const float*)d_in[2];   // (8192, 256)
  const float* cp = (const float*)d_in[3];   // (256,)
  float* out = (float*)d_out;

  char* ws = (char*)d_ws;
  unsigned short* prT = (unsigned short*)ws;                  // 4 MB  bf16 256x8192
  float* Mmat = (float*)(ws + (size_t)(4u << 20));            // 8 MB  f32  8192x256
  float* rowW = (float*)(ws + (size_t)(12u << 20));           // 32 KB f32  8192

  // zero the atomic accumulators (Mmat + rowW are contiguous) and the output
  hipMemsetAsync(ws + (size_t)(4u << 20), 0, (size_t)(8u << 20) + A_DIM * sizeof(float), stream);
  hipMemsetAsync(out, 0, sizeof(float), stream);

  k_transpose_pr<<<dim3(B_DIM / 32, K_CL / 32), dim3(32, 8), 0, stream>>>(pR, prT);
  k_gemm_wpr<<<dim3(A_DIM / BM, KSPLIT), 256, 0, stream>>>(W, prT, Mmat, rowW);
  k_finalize<<<dim3(A_DIM / 128), 256, 0, stream>>>(pL, Mmat, rowW, cp, out);
}
